// DecoderBlock_49804440764760
// MI455X (gfx1250) — matrix operand order repaired and checked by reading, not compiled, not run on hardware
//
#include <hip/hip_runtime.h>

typedef __attribute__((ext_vector_type(16))) _Float16 v16h;
typedef __attribute__((ext_vector_type(8)))  float    v8f;
typedef _Float16 F16;

union F16x16 { uint4 u[2]; v16h v; F16 h[16]; };

// ---------------- small helper kernels ----------------

__global__ void k_f32_to_f16(const float* __restrict__ src, F16* __restrict__ dst, int n) {
  int i = blockIdx.x * 256 + threadIdx.x;
  if (i < n) dst[i] = (F16)src[i];
}

// decoder conv weights: src [M][512][9] (OIHW) -> dst [M][9][512] (p-major K ordering, f16)
__global__ void k_conv_w(const float* __restrict__ src, F16* __restrict__ dst, int M) {
  int idx = blockIdx.x * 256 + threadIdx.x;
  if (idx >= M * 4608) return;
  int m = idx / 4608, r = idx - m * 4608;
  int p = r >> 9, ci = r & 511;
  dst[idx] = (F16)src[(size_t)(m * 512 + ci) * 9 + p];
}

// pooled[b,s] = mean over 3x3 of style map w
__global__ void k_pooled(const float* __restrict__ w, float* __restrict__ pooled) {
  int i = blockIdx.x * 256 + threadIdx.x;           // 0..8191 = b*512+s
  if (i < 16 * 512) {
    const float* p = w + (size_t)i * 9;
    float s = 0.f;
#pragma unroll
    for (int j = 0; j < 9; ++j) s += p[j];
    pooled[i] = s * (1.f / 9.f);
  }
}

// pwbuf[b,4096] = pooled@kp_pw^T + kp_pb ; biasbuf[b,512] = pooled@kp_bw^T + kp_bb
__global__ void k_pw_bias(const float* __restrict__ pooled,
                          const float* __restrict__ kp_pw, const float* __restrict__ kp_pb,
                          const float* __restrict__ kp_bw, const float* __restrict__ kp_bb,
                          float* __restrict__ pwbuf, float* __restrict__ biasbuf) {
  int b = blockIdx.y;
  int j = blockIdx.x * 256 + threadIdx.x;           // 0..4607
  __shared__ float ps[512];
  ps[threadIdx.x]       = pooled[b * 512 + threadIdx.x];
  ps[threadIdx.x + 256] = pooled[b * 512 + threadIdx.x + 256];
  __syncthreads();
  const float* row; float bb; float* dst;
  if (j < 4096) { row = kp_pw + (size_t)j * 512; bb = kp_pb[j]; dst = pwbuf + b * 4096 + j; }
  else { int c = j - 4096; row = kp_bw + (size_t)c * 512; bb = kp_bb[c]; dst = biasbuf + b * 512 + c; }
  float s = bb;
  for (int k = 0; k < 512; ++k) s += ps[k] * row[k];
  *dst = s;
}

__device__ __forceinline__ int refl5(int r) {       // reflect-pad(1) index into size-3 axis
  int i = r - 1;
  return i < 0 ? -i : (i > 2 ? 4 - i : i);
}

// BmatT[col=b*16+p][k=s*9+i*3+j] = reflect-padded style patch (p<9 valid, else 0)
__global__ void k_bmat(const float* __restrict__ w, F16* __restrict__ BmatT) {
  int idx = blockIdx.x * 256 + threadIdx.x;         // 256*4608 total
  if (idx >= 256 * 4608) return;
  int col = idx / 4608, k = idx % 4608;
  int b = col >> 4, p = col & 15;
  F16 val = (F16)0.f;
  if (p < 9) {
    int kh = p / 3, kw = p % 3;
    int s = k / 9, q = k % 9, i = q / 3, j = q % 3;
    val = (F16)w[((b * 512 + s) * 3 + refl5(kh + i)) * 3 + refl5(kw + j)];
  }
  BmatT[idx] = val;
}

// instance-norm statistics per (b,c) plane of 1024 pixels
__global__ void k_instats(const float* __restrict__ x, float* __restrict__ mean,
                          float* __restrict__ rstd) {
  int pc = blockIdx.x;                              // 0..8191
  const float* p = x + (size_t)pc * 1024;
  float s = 0.f, s2 = 0.f;
  for (int i = threadIdx.x; i < 1024; i += 256) { float v = p[i]; s += v; s2 += v * v; }
  __shared__ float sh1[256], sh2[256];
  sh1[threadIdx.x] = s; sh2[threadIdx.x] = s2;
  __syncthreads();
  for (int off = 128; off > 0; off >>= 1) {
    if (threadIdx.x < off) { sh1[threadIdx.x] += sh1[threadIdx.x + off];
                             sh2[threadIdx.x] += sh2[threadIdx.x + off]; }
    __syncthreads();
  }
  if (threadIdx.x == 0) {
    float m = sh1[0] * (1.f / 1024.f);
    float v = sh2[0] * (1.f / 1024.f) - m * m;
    mean[pc] = m; rstd[pc] = rsqrtf(v + 1e-5f);
  }
}

// W_eff[b][g][o][i][p] = sum_m pw[b,g,o,m] * dw[b, g*8+m, i, p]  (folds pointwise into spatial)
__global__ void k_weff(const float* __restrict__ pwbuf, const float* __restrict__ dw,
                       float* __restrict__ weff) {
  int idx = blockIdx.x * 256 + threadIdx.x;
  if (idx >= 16 * 64 * 8 * 8 * 9) return;
  int p = idx % 9; int r = idx / 9;
  int i = r & 7; r >>= 3;
  int o = r & 7; r >>= 3;
  int g = r & 63; int b = r >> 6;
  const float* pw = pwbuf + (size_t)b * 4096 + g * 64 + o * 8;
  float s = 0.f;
#pragma unroll
  for (int m = 0; m < 8; ++m)
    s += pw[m] * dw[((size_t)b * 4096 + (g * 8 + m) * 8 + i) * 9 + p];
  weff[idx] = s;
}

// dynamic grouped conv: zpad[b][c][1+h][1+w] = IN(x) (*) W_eff + bias  (reflect pad, K=72)
__global__ __launch_bounds__(256)
void k_staged(const float* __restrict__ x, const float* __restrict__ mean,
              const float* __restrict__ rstd, const float* __restrict__ weff,
              const float* __restrict__ biasbuf, F16* __restrict__ zpad) {
  int b = blockIdx.z, g = blockIdx.y;
  int h0 = (blockIdx.x >> 1) * 16, w0 = (blockIdx.x & 1) * 16;
  __shared__ float xs[8][18][18];
  __shared__ float ws[8][72];
  __shared__ float bs[8];
  int t = threadIdx.x;
  for (int idx = t; idx < 8 * 18 * 18; idx += 256) {
    int ch = idx / 324, r = idx % 324, hh = r / 18, ww = r % 18;
    int gh = h0 - 1 + hh; gh = gh < 0 ? -gh : (gh > 31 ? 62 - gh : gh);
    int gw = w0 - 1 + ww; gw = gw < 0 ? -gw : (gw > 31 ? 62 - gw : gw);
    int c = b * 512 + g * 8 + ch;
    xs[ch][hh][ww] = (x[((size_t)c << 10) + (gh << 5) + gw] - mean[c]) * rstd[c];
  }
  for (int idx = t; idx < 576; idx += 256)
    ws[idx / 72][idx % 72] = weff[((size_t)b * 64 + g) * 576 + idx];
  if (t < 8) bs[t] = biasbuf[b * 512 + g * 8 + t];
  __syncthreads();
  int o = t >> 5, lane = t & 31;
#pragma unroll
  for (int j = 0; j < 8; ++j) {
    int px = lane + j * 32;
    int hy = px >> 4, wx = px & 15;
    float s = bs[o];
    for (int i = 0; i < 8; ++i)
#pragma unroll
      for (int kh = 0; kh < 3; ++kh)
#pragma unroll
        for (int kw = 0; kw < 3; ++kw)
          s += xs[i][hy + kh][wx + kw] * ws[o][i * 9 + kh * 3 + kw];
    zpad[(((size_t)b * 512 + g * 8 + o) * 34 + hy + h0 + 1) * 34 + (wx + w0 + 1)] = (F16)s;
  }
}

// ---------------- WMMA implicit-GEMM kernel ----------------
// Block tile 128M x 128N x 32K, 256 threads = 8 waves (4M x 2N), wave tile 32x64.
// Double-buffered LDS, single barrier per K-step. A-tile prefetch is stored into the
// idle buffer at the top of the iteration (hazard-free); only the 8-VGPR im2col B
// fragment stays live across the WMMA burst.
// Conv modes use K ordered as k = p*512 + ci (p-major): a 32-wide K-step never
// crosses a tap boundary, so per-thread im2col addresses are affine (base + kk*1156).
enum { MODE_DENSE = 0, MODE_CONV_PAD = 1, MODE_CONV_UP = 2 };

template <int MODE>
__global__ __launch_bounds__(256, 2)
void k_gemm_wmma(const F16* __restrict__ A, const F16* __restrict__ Bd,
                 const F16* __restrict__ srcpad, const float* __restrict__ bias,
                 void* __restrict__ dstv, int K) {
  __shared__ alignas(16) F16 As[2][128][32];
  __shared__ alignas(16) F16 Bs[2][128][32];
  const int t = threadIdx.x;
  const int lane = t & 31, wave = t >> 5;
  const int wm = wave >> 1, wn = wave & 1;
  const int m0 = blockIdx.y * 128;
  const int n0 = blockIdx.x * 128;

  v8f zero = {};
  v8f acc[2][4];
#pragma unroll
  for (int ms = 0; ms < 2; ++ms)
#pragma unroll
    for (int s = 0; s < 4; ++s) acc[ms][s] = zero;

  // per-thread staging coordinates: A tile 128x32 halves, B tile 128x32 halves
  const int a_m = t >> 1, a_c = (t & 1) * 16;       // 2 uint4 per thread
  int b_n, b_kc;
  const F16* bbase = nullptr;                        // per-thread spatial base (conv modes)
  if constexpr (MODE == MODE_DENSE) { b_n = t >> 1; b_kc = (t & 1) * 16; }
  else {
    b_n = t & 127; b_kc = (t >> 7) * 16;
    int n = n0 + b_n;
    int cb = n >> 10, r = n & 1023, ch = r >> 5, cw = r & 31;
    bbase = srcpad + ((size_t)cb * 512 * 34 + ch) * 34 + cw;
  }
  const int hi = lane >> 4;

  auto load_a_store = [&](int buf, int k0) {        // A: load + stage immediately
    const uint4* asrc = reinterpret_cast<const uint4*>(A + (size_t)(m0 + a_m) * K + k0 + a_c);
    uint4 a0 = asrc[0], a1 = asrc[1];
    *reinterpret_cast<uint4*>(&As[buf][a_m][a_c])     = a0;
    *reinterpret_cast<uint4*>(&As[buf][a_m][a_c + 8]) = a1;
  };
  auto load_b = [&](int k0) -> F16x16 {             // B: issue loads, keep in regs
    F16x16 r;
    if constexpr (MODE == MODE_DENSE) {
      const uint4* bsrc = reinterpret_cast<const uint4*>(Bd + (size_t)(n0 + b_n) * K + k0 + b_kc);
      r.u[0] = bsrc[0]; r.u[1] = bsrc[1];
    } else {
      int kq = k0 + b_kc;                            // k = p*512 + ci
      int pp = kq >> 9, ci0 = kq & 511;
      int kh = pp / 3, kw = pp - kh * 3;
      const F16* bp = bbase + (size_t)ci0 * 1156 + kh * 34 + kw;
#pragma unroll
      for (int kk = 0; kk < 16; ++kk)                // affine: immediate-offset b16 loads
        r.h[kk] = bp[kk * 1156];
    }
    return r;
  };
  auto store_b = [&](int buf, const F16x16& r) {
    *reinterpret_cast<uint4*>(&Bs[buf][b_n][b_kc])     = r.u[0];
    *reinterpret_cast<uint4*>(&Bs[buf][b_n][b_kc + 8]) = r.u[1];
  };

  // prologue: stage tile 0
  load_a_store(0, 0);
  store_b(0, load_b(0));
  __syncthreads();

  const int nk = K >> 5;
  for (int i = 0; i < nk; ++i) {
    const int cur = i & 1;
    const bool more = (i + 1 < nk);
    F16x16 nb;
    if (more) {
      int k0n = (i + 1) << 5;
      nb = load_b(k0n);                              // global loads in flight during WMMAs
      load_a_store(cur ^ 1, k0n);                    // next buffer is idle this iteration
    }

    // A fragments: lane<16 -> K {0..7,16..23}; lane>=16 -> K {8..15,24..31}
    F16x16 af[2];
#pragma unroll
    for (int ms = 0; ms < 2; ++ms) {
      const uint4* arow =
          reinterpret_cast<const uint4*>(&As[cur][wm * 32 + ms * 16 + (lane & 15)][0]);
      af[ms].u[0] = arow[hi];
      af[ms].u[1] = arow[2 + hi];
    }
#pragma unroll
    for (int s = 0; s < 4; ++s) {
      // B fragment: lane<16 -> K {0..7,16..23}; lane>=16 -> K {8..15,24..31} (column = lane&15)
      F16x16 bf;
      const uint4* brow =
          reinterpret_cast<const uint4*>(&Bs[cur][wn * 64 + s * 16 + (lane & 15)][0]);
      bf.u[0] = brow[hi];
      bf.u[1] = brow[2 + hi];
      acc[0][s] = __builtin_amdgcn_wmma_f32_16x16x32_f16(false, af[0].v, false, bf.v,
                                                         (short)0, acc[0][s], false, false);
      acc[1][s] = __builtin_amdgcn_wmma_f32_16x16x32_f16(false, af[1].v, false, bf.v,
                                                         (short)0, acc[1][s], false, false);
    }
    if (more) store_b(cur ^ 1, nb);
    __syncthreads();
  }

  // epilogue: C layout -> row = m-base + v + 8*(lane>=16), col = lane&15
#pragma unroll
  for (int ms = 0; ms < 2; ++ms) {
#pragma unroll
    for (int s = 0; s < 4; ++s) {
      int col = n0 + wn * 64 + s * 16 + (lane & 15);
#pragma unroll
      for (int v = 0; v < 8; ++v) {
        int row = m0 + wm * 32 + ms * 16 + v + hi * 8;
        float val = acc[ms][s][v];
        if constexpr (MODE == MODE_DENSE) {
          int b = col >> 4, p = col & 15;
          if (p < 9) {
            float* dst = (float*)dstv;              // dw[b][4096][9]
            dst[((size_t)b * 4096 + row) * 9 + p] = val + bias[row];
          }
        } else if constexpr (MODE == MODE_CONV_PAD) {
          val = fmaxf(val + bias[row], 0.f);
          int b = col >> 10, r = col & 1023, h = r >> 5, w = r & 31;
          F16* dst = (F16*)dstv;                    // y1pad[b][512][34][34]
          dst[(((size_t)b * 512 + row) * 34 + h + 1) * 34 + (w + 1)] = (F16)val;
        } else {
          val = fmaxf(val + bias[row], 0.f);
          int b = col >> 10, r = col & 1023, h = r >> 5, w = r & 31;
          float* dst = (float*)dstv;                // out[b][256][64][64], 2x2 nearest
          size_t base = (((size_t)b * 256 + row) * 64 + 2 * h) * 64 + 2 * w;
          float2 vv; vv.x = val; vv.y = val;
          *reinterpret_cast<float2*>(dst + base)      = vv;   // b64 store
          *reinterpret_cast<float2*>(dst + base + 64) = vv;   // b64 store
        }
      }
    }
  }
}

// ---------------- launcher ----------------

extern "C" void kernel_launch(void* const* d_in, const int* in_sizes, int n_in,
                              void* d_out, int out_size, void* d_ws, size_t ws_size,
                              hipStream_t stream) {
  (void)in_sizes; (void)n_in; (void)out_size; (void)ws_size;
  const float* x      = (const float*)d_in[0];
  const float* w      = (const float*)d_in[1];
  const float* kp_sw  = (const float*)d_in[2];
  const float* kp_sb  = (const float*)d_in[3];
  const float* kp_pw  = (const float*)d_in[4];
  const float* kp_pb  = (const float*)d_in[5];
  const float* kp_bw  = (const float*)d_in[6];
  const float* kp_bb  = (const float*)d_in[7];
  const float* dec_w1 = (const float*)d_in[8];
  const float* dec_b1 = (const float*)d_in[9];
  const float* dec_w2 = (const float*)d_in[10];
  const float* dec_b2 = (const float*)d_in[11];
  float* out = (float*)d_out;

  char* ws = (char*)d_ws;
  size_t off = 0;
  auto alloc = [&](size_t bytes) -> char* {
    char* p = ws + off; off += (bytes + 255) & ~size_t(255); return p;
  };
  F16*   A1      = (F16*)alloc((size_t)4096 * 4608 * 2);   // kp_sw f16 (s-major K)
  F16*   A2      = (F16*)alloc((size_t)512 * 4608 * 2);    // dec_w1 f16 (p-major K)
  F16*   A3      = (F16*)alloc((size_t)256 * 4608 * 2);    // dec_w2 f16 (p-major K)
  F16*   BT      = (F16*)alloc((size_t)256 * 4608 * 2);    // style patches, transposed
  float* dwb     = (float*)alloc((size_t)16 * 4096 * 9 * 4);
  float* pooled  = (float*)alloc(8192 * 4);
  float* pwbuf   = (float*)alloc((size_t)16 * 4096 * 4);
  float* biasbuf = (float*)alloc(16 * 512 * 4);
  float* meanb   = (float*)alloc(8192 * 4);
  float* rstdb   = (float*)alloc(8192 * 4);
  float* weff    = (float*)alloc((size_t)16 * 64 * 576 * 4);
  F16*   zpad    = (F16*)alloc((size_t)16 * 512 * 34 * 34 * 2);
  F16*   y1pad   = (F16*)alloc((size_t)16 * 512 * 34 * 34 * 2);

  // weight conversions for WMMA (decoder weights permuted to p-major K)
  k_f32_to_f16<<<(4096 * 4608 + 255) / 256, 256, 0, stream>>>(kp_sw, A1, 4096 * 4608);
  k_conv_w<<<(512 * 4608 + 255) / 256, 256, 0, stream>>>(dec_w1, A2, 512);
  k_conv_w<<<(256 * 4608 + 255) / 256, 256, 0, stream>>>(dec_w2, A3, 256);

  // style pooling + pointwise/bias predictors
  k_pooled<<<32, 256, 0, stream>>>(w, pooled);
  k_pw_bias<<<dim3(18, 16), 256, 0, stream>>>(pooled, kp_pw, kp_pb, kp_bw, kp_bb, pwbuf, biasbuf);

  // spatial kernel predictor as one WMMA GEMM (A shared across batch), M=4096 N=256 K=4608
  k_bmat<<<(256 * 4608) / 256, 256, 0, stream>>>(w, BT);
  k_gemm_wmma<MODE_DENSE><<<dim3(2, 32), 256, 0, stream>>>(A1, BT, nullptr, kp_sb, dwb, 4608);
  k_weff<<<(16 * 64 * 8 * 8 * 9 + 255) / 256, 256, 0, stream>>>(pwbuf, dwb, weff);

  // instance norm stats, then fused dynamic grouped conv into zero-padded f16 tensor
  k_instats<<<8192, 256, 0, stream>>>(x, meanb, rstdb);
  hipMemsetAsync(zpad, 0, (size_t)16 * 512 * 34 * 34 * 2, stream);
  hipMemsetAsync(y1pad, 0, (size_t)16 * 512 * 34 * 34 * 2, stream);
  k_staged<<<dim3(4, 64, 16), 256, 0, stream>>>(x, meanb, rstdb, weff, biasbuf, zpad);

  // decoder convs as WMMA implicit GEMMs (M=512/256, N=16384, K=4608)
  k_gemm_wmma<MODE_CONV_PAD><<<dim3(128, 4), 256, 0, stream>>>(A2, nullptr, zpad, dec_b1, y1pad, 4608);
  k_gemm_wmma<MODE_CONV_UP><<<dim3(128, 2), 256, 0, stream>>>(A3, nullptr, y1pad, dec_b2, out, 4608);
}
